// Attention_57458072486316
// MI455X (gfx1250) — compile-verified, matrix operand order checked by reading, not run on hardware
//
#include <hip/hip_runtime.h>
#include <hip/hip_bf16.h>

// ---------------------------------------------------------------------------
// CDNA5 (gfx1250, wave32) fused MHA: bf16 WMMA pipeline.
// ---------------------------------------------------------------------------

typedef __attribute__((ext_vector_type(16))) __bf16 v16bf;
typedef __attribute__((ext_vector_type(8)))  __bf16 v8bf;
typedef __attribute__((ext_vector_type(4)))  __bf16 v4bf;
typedef __attribute__((ext_vector_type(8)))  float  v8f;

#define D_MODEL 1024
#define HEADS   16
#define D_H     64

union V16U { v16bf v; v8bf h[2]; };

// Load one 16x32 bf16 operand tile in the CDNA5 WMMA A/B striping:
// lane L owns row (L&15); its 16 elements are two 16B chunks at
// k-offsets 8*(L>>4) and 16+8*(L>>4).
__device__ __forceinline__ v16bf load_tile16x32(const __bf16* base, int ld) {
    const int lane = threadIdx.x & 31;
    const int r    = lane & 15;
    const int hh   = lane >> 4;
    const __bf16* rowp = base + (size_t)r * ld;
    V16U u;
    u.h[0] = *(const v8bf*)(rowp + hh * 8);
    u.h[1] = *(const v8bf*)(rowp + 16 + hh * 8);
    return u.v;
}

__device__ __forceinline__ v8f wmma_bf16(v16bf a, v16bf b, v8f c) {
    return __builtin_amdgcn_wmma_f32_16x16x32_bf16(
        false, a, false, b, (short)0, c, false, false);
}

// ---- DPP 16-lane butterfly reductions (pure VALU, no LDS, no waits) --------
template <int CTRL>
__device__ __forceinline__ float dpp_movf(float x) {
    int xi = __builtin_bit_cast(int, x);
    int r  = __builtin_amdgcn_update_dpp(xi, xi, CTRL, 0xF, 0xF, true);
    return __builtin_bit_cast(float, r);
}
// quad_perm(1,0,3,2)=0xB1, quad_perm(2,3,0,1)=0x4E,
// row_half_mirror=0x141, row_mirror=0x140 (DPP row = 16 lanes)
__device__ __forceinline__ float row16_max(float x) {
    x = fmaxf(x, dpp_movf<0xB1>(x));
    x = fmaxf(x, dpp_movf<0x4E>(x));
    x = fmaxf(x, dpp_movf<0x141>(x));
    x = fmaxf(x, dpp_movf<0x140>(x));
    return x;
}
__device__ __forceinline__ float row16_sum(float x) {
    x += dpp_movf<0xB1>(x);
    x += dpp_movf<0x4E>(x);
    x += dpp_movf<0x141>(x);
    x += dpp_movf<0x140>(x);
    return x;
}

// ---------------------------------------------------------------------------
// Stage 0: f32 -> bf16 conversion (vectorized, 4 elems/lane)
// ---------------------------------------------------------------------------
__global__ void cvt_f32_bf16(const float* __restrict__ in,
                             __bf16* __restrict__ out, int n) {
    int i = (blockIdx.x * blockDim.x + threadIdx.x) * 4;
    if (i < n) {
        float4 f = *(const float4*)(in + i);
        v4bf o;
        o[0] = (__bf16)f.x; o[1] = (__bf16)f.y;
        o[2] = (__bf16)f.z; o[3] = (__bf16)f.w;
        *(v4bf*)(out + i) = o;
    }
}

// ---------------------------------------------------------------------------
// Stage 1/3: GEMM  Y = (X @ W^T + b) * oscale, software-pipelined K loop.
// Block = 128 threads (4 waves); block tile 16x256, wave tile 16x64.
// MODE 0: out bf16 [H][S][64] | MODE 1: out bf16 [H][64][S] | MODE 2: f32 [S,1024]
// ---------------------------------------------------------------------------
template <int MODE>
__global__ __launch_bounds__(128, 1)
void gemm_bf16(const __bf16* __restrict__ X,
               const __bf16* __restrict__ W,
               const float* __restrict__ bias,
               void* __restrict__ out, int S, float oscale) {
    const int K    = D_MODEL;
    const int wave = threadIdx.x >> 5;
    const int lane = threadIdx.x & 31;
    const int row0 = blockIdx.x * 16;
    const int col0 = blockIdx.y * 256 + wave * 64;

    v8f acc[4] = {};
    const __bf16* xrow = X + (size_t)row0 * K;
    const __bf16* w0 = W + (size_t)(col0 +  0) * K;
    const __bf16* w1 = W + (size_t)(col0 + 16) * K;
    const __bf16* w2 = W + (size_t)(col0 + 32) * K;
    const __bf16* w3 = W + (size_t)(col0 + 48) * K;

    // prologue loads
    v16bf a  = load_tile16x32(xrow, K);
    v16bf b0 = load_tile16x32(w0, K);
    v16bf b1 = load_tile16x32(w1, K);
    v16bf b2 = load_tile16x32(w2, K);
    v16bf b3 = load_tile16x32(w3, K);

    for (int kk = 0; kk < K; kk += 32) {
        const int kn = (kk + 32 < K) ? kk + 32 : kk;   // clamped prefetch
        v16bf an  = load_tile16x32(xrow + kn, K);
        v16bf bn0 = load_tile16x32(w0 + kn, K);
        v16bf bn1 = load_tile16x32(w1 + kn, K);
        v16bf bn2 = load_tile16x32(w2 + kn, K);
        v16bf bn3 = load_tile16x32(w3 + kn, K);
        acc[0] = wmma_bf16(a, b0, acc[0]);
        acc[1] = wmma_bf16(a, b1, acc[1]);
        acc[2] = wmma_bf16(a, b2, acc[2]);
        acc[3] = wmma_bf16(a, b3, acc[3]);
        a = an; b0 = bn0; b1 = bn1; b2 = bn2; b3 = bn3;
    }

    const int hh = lane >> 4, nl = lane & 15;
    #pragma unroll
    for (int j = 0; j < 4; ++j) {
        #pragma unroll
        for (int v = 0; v < 8; ++v) {
            int m = row0 + v + 8 * hh;
            int n = col0 + j * 16 + nl;
            float val = (acc[j][v] + bias[n]) * oscale;
            if (MODE == 0) {
                int hd = n >> 6, d = n & 63;
                ((__bf16*)out)[((size_t)hd * S + m) * D_H + d] = (__bf16)val;
            } else if (MODE == 1) {
                int hd = n >> 6, d = n & 63;
                ((__bf16*)out)[((size_t)hd * D_H + d) * S + m] = (__bf16)val;
            } else {
                ((float*)out)[(size_t)m * D_MODEL + n] = val;
            }
        }
    }
}

// ---------------------------------------------------------------------------
// Stage 2: flash attention.  One wave = one 16-query tile of one head.
// 32 keys/iteration, software-pipelined: next K-operands load during the
// softmax of the current tile; V-operands issue before the softmax so they
// are resident for the P*V WMMAs.  Q pre-scaled by 1/sqrt(d_h).
// Q,K: [H][S][64] bf16;  Vt: [H][64][S] bf16;  ctx: [S,1024] bf16.
// ---------------------------------------------------------------------------
__global__ __launch_bounds__(128, 1)
void attn_fwd(const __bf16* __restrict__ Q,
              const __bf16* __restrict__ Km,
              const __bf16* __restrict__ Vt,
              const int* __restrict__ mask,
              __bf16* __restrict__ ctx, int S) {
    __shared__ __bf16 pbuf[4][16 * 32];   // per-wave P staging tile (1KB)

    const int wave = threadIdx.x >> 5;
    const int lane = threadIdx.x & 31;
    const int head = blockIdx.y;
    const int q0   = (blockIdx.x * 4 + wave) * 16;
    const int hh   = lane >> 4, ln = lane & 15;

    const __bf16* Qh = Q  + ((size_t)head * S + q0) * D_H;
    const __bf16* Kh = Km + (size_t)head * S * D_H;
    const __bf16* Vh = Vt + (size_t)head * D_H * S;

    // Q tile: 16 x 64 = two A operands, loaded once (already scaled).
    v16bf aq0 = load_tile16x32(Qh + 0,  D_H);
    v16bf aq1 = load_tile16x32(Qh + 32, D_H);

    v8f o[4] = {};
    float mrun[8], lrun[8];
    #pragma unroll
    for (int v = 0; v < 8; ++v) { mrun[v] = -1e30f; lrun[v] = 0.f; }

    __bf16* myp = &pbuf[wave][0];

    // prologue: K operands for the first 32-key block
    v16bf kb0 = load_tile16x32(Kh + 0 * D_H + 0,  D_H);
    v16bf kb1 = load_tile16x32(Kh + 0 * D_H + 32, D_H);
    v16bf kb2 = load_tile16x32(Kh + 16 * D_H + 0,  D_H);
    v16bf kb3 = load_tile16x32(Kh + 16 * D_H + 32, D_H);

    for (int kt = 0; kt < S; kt += 32) {
        // V operands for this block: issue early, consumed after softmax
        v16bf bv0 = load_tile16x32(Vh + (size_t) 0 * S + kt, S);
        v16bf bv1 = load_tile16x32(Vh + (size_t)16 * S + kt, S);
        v16bf bv2 = load_tile16x32(Vh + (size_t)32 * S + kt, S);
        v16bf bv3 = load_tile16x32(Vh + (size_t)48 * S + kt, S);

        // --- scores for 32 keys: two 16x16 tiles ---
        v8f s0 = {}, s1 = {};
        s0 = wmma_bf16(aq0, kb0, s0);
        s0 = wmma_bf16(aq1, kb1, s0);
        s1 = wmma_bf16(aq0, kb2, s1);
        s1 = wmma_bf16(aq1, kb3, s1);

        // prefetch next block's K operands (clamped on last iteration);
        // these overlap the whole softmax stage below.
        const int ktn = (kt + 32 < S) ? kt + 32 : kt;
        kb0 = load_tile16x32(Kh + (size_t)(ktn +  0) * D_H + 0,  D_H);
        kb1 = load_tile16x32(Kh + (size_t)(ktn +  0) * D_H + 32, D_H);
        kb2 = load_tile16x32(Kh + (size_t)(ktn + 16) * D_H + 0,  D_H);
        kb3 = load_tile16x32(Kh + (size_t)(ktn + 16) * D_H + 32, D_H);

        // key mask (scale already folded into Q projection)
        const int mk0 = mask[kt + ln];
        const int mk1 = mask[kt + 16 + ln];
        #pragma unroll
        for (int v = 0; v < 8; ++v) {
            s0[v] = mk0 ? s0[v] : -1e9f;
            s1[v] = mk1 ? s1[v] : -1e9f;
        }

        // --- online softmax: DPP butterfly reductions per row ---
        #pragma unroll
        for (int v = 0; v < 8; ++v) {
            float loc   = row16_max(fmaxf(s0[v], s1[v]));
            float mnew  = fmaxf(mrun[v], loc);
            float alpha = __expf(mrun[v] - mnew);
            float e0 = __expf(s0[v] - mnew);
            float e1 = __expf(s1[v] - mnew);
            float rs = row16_sum(e0 + e1);
            lrun[v] = lrun[v] * alpha + rs;
            mrun[v] = mnew;
            #pragma unroll
            for (int j = 0; j < 4; ++j) o[j][v] *= alpha;

            // P -> LDS row-major [16][32] (query row, key-local col)
            int mrow = v + 8 * hh;
            myp[mrow * 32 +      ln] = (__bf16)e0;
            myp[mrow * 32 + 16 + ln] = (__bf16)e1;
        }

        // --- P (16x32 via LDS, A-layout) x V (32x64) ---
        v16bf ap = load_tile16x32(myp, 32);
        o[0] = wmma_bf16(ap, bv0, o[0]);
        o[1] = wmma_bf16(ap, bv1, o[1]);
        o[2] = wmma_bf16(ap, bv2, o[2]);
        o[3] = wmma_bf16(ap, bv3, o[3]);
    }

    // normalize + write context [S, 1024] bf16
    float inv[8];
    #pragma unroll
    for (int v = 0; v < 8; ++v) inv[v] = __builtin_amdgcn_rcpf(lrun[v]);
    #pragma unroll
    for (int j = 0; j < 4; ++j) {
        #pragma unroll
        for (int v = 0; v < 8; ++v) {
            int m = q0 + v + 8 * hh;
            int n = head * D_H + j * 16 + ln;
            ctx[(size_t)m * D_MODEL + n] = (__bf16)(o[j][v] * inv[v]);
        }
    }
}

// ---------------------------------------------------------------------------
// Launcher
// ---------------------------------------------------------------------------
extern "C" void kernel_launch(void* const* d_in, const int* in_sizes, int n_in,
                              void* d_out, int out_size, void* d_ws, size_t ws_size,
                              hipStream_t stream) {
    const float* q_in = (const float*)d_in[0];
    const float* k_in = (const float*)d_in[1];
    const float* v_in = (const float*)d_in[2];
    const int*   mask = (const int*)  d_in[3];
    const float* Wq = (const float*)d_in[4];  const float* bq = (const float*)d_in[5];
    const float* Wk = (const float*)d_in[6];  const float* bk = (const float*)d_in[7];
    const float* Wv = (const float*)d_in[8];  const float* bv = (const float*)d_in[9];
    const float* Wo = (const float*)d_in[10]; const float* bo = (const float*)d_in[11];
    float* out = (float*)d_out;

    const int S  = in_sizes[0] / D_MODEL;     // 4096
    const int NX = S * D_MODEL;
    const int NW = D_MODEL * D_MODEL;

    __bf16* ws  = (__bf16*)d_ws;
    __bf16* Xq  = ws;
    __bf16* Xk  = Xq  + NX;
    __bf16* Xv  = Xk  + NX;
    __bf16* Wqb = Xv  + NX;
    __bf16* Wkb = Wqb + NW;
    __bf16* Wvb = Wkb + NW;
    __bf16* Wob = Wvb + NW;
    __bf16* Qh  = Wob + NW;           // [16][S][64]  (pre-scaled by 1/8)
    __bf16* Kh  = Qh  + NX;
    __bf16* Vt  = Kh  + NX;           // [16][64][S]
    __bf16* ctx = Vt  + NX;           // [S,1024]

    {
        int tb = 256;
        cvt_f32_bf16<<<NX / (4 * tb), tb, 0, stream>>>(q_in, Xq, NX);
        cvt_f32_bf16<<<NX / (4 * tb), tb, 0, stream>>>(k_in, Xk, NX);
        cvt_f32_bf16<<<NX / (4 * tb), tb, 0, stream>>>(v_in, Xv, NX);
        cvt_f32_bf16<<<NW / (4 * tb), tb, 0, stream>>>(Wq, Wqb, NW);
        cvt_f32_bf16<<<NW / (4 * tb), tb, 0, stream>>>(Wk, Wkb, NW);
        cvt_f32_bf16<<<NW / (4 * tb), tb, 0, stream>>>(Wv, Wvb, NW);
        cvt_f32_bf16<<<NW / (4 * tb), tb, 0, stream>>>(Wo, Wob, NW);
    }

    dim3 ggrid(S / 16, D_MODEL / 256);
    // Q projection carries the 1/sqrt(d_h)=0.125 attention scale (exact in bf16)
    gemm_bf16<0><<<ggrid, 128, 0, stream>>>(Xq, Wqb, bq, (void*)Qh, S, 0.125f);
    gemm_bf16<0><<<ggrid, 128, 0, stream>>>(Xk, Wkb, bk, (void*)Kh, S, 1.0f);
    gemm_bf16<1><<<ggrid, 128, 0, stream>>>(Xv, Wvb, bv, (void*)Vt, S, 1.0f);

    dim3 agrid(S / 64, HEADS);
    attn_fwd<<<agrid, 128, 0, stream>>>(Qh, Kh, Vt, mask, ctx, S);

    gemm_bf16<2><<<ggrid, 128, 0, stream>>>(ctx, Wob, bo, (void*)out, S, 1.0f);
}
